// Attention_60043642798184
// MI455X (gfx1250) — matrix operand order repaired and checked by reading, not compiled, not run on hardware
//
#include <hip/hip_runtime.h>
#include <hip/hip_bf16.h>

typedef __attribute__((ext_vector_type(16))) _Float16 v16h;
typedef __attribute__((ext_vector_type(8)))  _Float16 v8h;
typedef __attribute__((ext_vector_type(8)))  float    v8f;

#define BATCH 2
#define SEQ   2048
#define CDIM  768
#define HEADS 12
#define HDIM  64

__device__ __forceinline__ v8f wmma_f16(v16h a, v16h b, v8f c) {
  // D = A(16x32 f16) * B(32x16 f16) + C(16x16 f32)
  return __builtin_amdgcn_wmma_f32_16x16x32_f16(false, a, false, b, (short)0, c,
                                                false, false);
}

// Build a 16-element fragment from two contiguous, 16B-aligned 8xf16 runs.
// Compiles to 2x ds_load_b128 instead of 16 scalar loads.
__device__ __forceinline__ v16h frag16(const _Float16* p0, const _Float16* p1) {
  v8h lo = *(const v8h*)p0;
  v8h hi = *(const v8h*)p1;
  return __builtin_shufflevector(lo, hi, 0, 1, 2, 3, 4, 5, 6, 7, 8, 9, 10, 11,
                                 12, 13, 14, 15);
}

// A-fragment (16x32 f16), tile stored row-major [16][stride], K contiguous.
// ISA layout: lane<16 row=lid, K = (e<8?e:e+8)+8*hl.
__device__ __forceinline__ v16h afrag(const _Float16* base, int stride, int lid,
                                      int hl, int koff) {
  const _Float16* p = base + lid * stride + koff + 8 * hl;
  return frag16(p, p + 16);
}

// B-fragment (32x16 f16), tile stored N-major [col][stride], K contiguous.
// ISA layout: K = (e<8?e:e+8)+8*hl.
__device__ __forceinline__ v16h bfrag(const _Float16* base, int stride, int col,
                                      int hl, int koff) {
  const _Float16* p = base + col * stride + koff + 8 * hl;
  return frag16(p, p + 16);
}

// ---------------------------------------------------------------------------
// Kernel 1: qkv = x @ w_qkv + b_qkv ; scatter to Q/K/V f16 [B,H,N,D], Q scaled.
// Block tile 128x64, 8 waves, 2x2 16x16 tiles per wave (8 WMMA / k-step).
// Grid: (2304/64, 4096/128) blocks of 256 threads. K-step 64.
// ---------------------------------------------------------------------------
__global__ __launch_bounds__(256) void qkv_gemm_kernel(
    const float* __restrict__ x, const float* __restrict__ w,
    const float* __restrict__ bias, _Float16* __restrict__ qf,
    _Float16* __restrict__ kf, _Float16* __restrict__ vf) {
  __shared__ _Float16 sA[128 * 72];  // [row][k] 128x64 (+8 pad)
  __shared__ _Float16 sB[64 * 72];   // [col][k] 64x64 (+8 pad), transposed
  const int t = threadIdx.x;
  const int wave = t >> 5, lane = t & 31, lid = lane & 15, hl = lane >> 4;
  const int m0 = blockIdx.y * 128;
  const int n0 = blockIdx.x * 64;
  const int pr = wave & 3;   // row pair: rows pr*32 .. pr*32+31
  const int pc = wave >> 2;  // col pair: cols pc*32 .. pc*32+31
  const int ar = t >> 1, ak = (t & 1) * 32;    // A stage: 2 thr/row, 32 f32
  const int col = t & 63, kc = (t >> 6) * 16;  // B stage: col-major transpose
  const v8f zero8 = {0.f, 0.f, 0.f, 0.f, 0.f, 0.f, 0.f, 0.f};
  v8f acc[2][2];
#pragma unroll
  for (int rr = 0; rr < 2; ++rr)
#pragma unroll
    for (int cc = 0; cc < 2; ++cc) acc[rr][cc] = zero8;

  for (int k0 = 0; k0 < CDIM; k0 += 64) {
    __syncthreads();
    {  // stage A, f32 -> f16, K contiguous
      const float4* src = (const float4*)(x + (size_t)(m0 + ar) * CDIM + k0 + ak);
#pragma unroll
      for (int q = 0; q < 4; ++q) {
        float4 f0 = src[2 * q], f1 = src[2 * q + 1];
        v8h h;
        h[0] = (_Float16)f0.x; h[1] = (_Float16)f0.y;
        h[2] = (_Float16)f0.z; h[3] = (_Float16)f0.w;
        h[4] = (_Float16)f1.x; h[5] = (_Float16)f1.y;
        h[6] = (_Float16)f1.z; h[7] = (_Float16)f1.w;
        *(v8h*)(sA + ar * 72 + ak + 8 * q) = h;
      }
      if (k0 + 64 < CDIM)
        __builtin_prefetch(x + (size_t)(m0 + ar) * CDIM + k0 + 64 + ak, 0, 3);
    }
    {  // stage B transposed: sB[col][k]; per-j reads are lane-coalesced
#pragma unroll
      for (int q = 0; q < 2; ++q) {
        v8h h;
#pragma unroll
        for (int j = 0; j < 8; ++j)
          h[j] = (_Float16)w[(size_t)(k0 + kc + 8 * q + j) * (3 * CDIM) + n0 + col];
        *(v8h*)(sB + col * 72 + kc + 8 * q) = h;
      }
      if (k0 + 64 < CDIM)
        __builtin_prefetch(w + (size_t)(k0 + 64 + kc) * (3 * CDIM) + n0 + col, 0, 3);
    }
    __syncthreads();
    v16h af[2][2], bf[2][2];
#pragma unroll
    for (int rr = 0; rr < 2; ++rr)
#pragma unroll
      for (int ks = 0; ks < 2; ++ks)
        af[rr][ks] = afrag(sA + (pr * 32 + rr * 16) * 72, 72, lid, hl, 32 * ks);
#pragma unroll
    for (int cc = 0; cc < 2; ++cc)
#pragma unroll
      for (int ks = 0; ks < 2; ++ks)
        bf[cc][ks] = bfrag(sB, 72, pc * 32 + cc * 16 + lid, hl, 32 * ks);
#pragma unroll
    for (int rr = 0; rr < 2; ++rr)
#pragma unroll
      for (int cc = 0; cc < 2; ++cc) {
        acc[rr][cc] = wmma_f16(af[rr][0], bf[cc][0], acc[rr][cc]);
        acc[rr][cc] = wmma_f16(af[rr][1], bf[cc][1], acc[rr][cc]);
      }
  }

  // Epilogue: every 16-wide col tile lies in one segment & one head; every
  // 16-tall row tile lies in one batch -> all splits are wave-uniform SALU.
#pragma unroll
  for (int rr = 0; rr < 2; ++rr) {
    const int rowbase = m0 + (pr * 2 + rr) * 16;
    const int bb = rowbase >> 11, ntok = rowbase & (SEQ - 1);
#pragma unroll
    for (int cc = 0; cc < 2; ++cc) {
      const int colbase = n0 + (pc * 2 + cc) * 16;
      const int seg = colbase / CDIM, cin = colbase % CDIM;
      const int hh = cin >> 6, dbase = cin & 63;
      _Float16* dst = (seg == 0) ? qf : (seg == 1 ? kf : vf);
      _Float16* tbase =
          dst + (((size_t)bb * HEADS + hh) * SEQ + ntok) * HDIM + dbase + lid;
      const float scale = (seg == 0) ? 0.125f : 1.0f;  // 1/sqrt(64) into Q
      const float bv = bias[colbase + lid];
#pragma unroll
      for (int r = 0; r < 8; ++r) {
        float v = (acc[rr][cc][r] + bv) * scale;
        tbase[(size_t)(r + 8 * hl) * HDIM] = (_Float16)v;
      }
    }
  }
}

// ---------------------------------------------------------------------------
// Kernel 2: flash attention. Grid: (B*H, SEQ/128), 256 threads (8 waves);
// wave w owns q rows [blockIdx.y*128 + 16w, +16). 32-key chunks, online
// softmax; scale folded into Q. K tile staged via async global->LDS (ASYNCcnt).
// ---------------------------------------------------------------------------
__global__ __launch_bounds__(256) void attn_kernel(
    const _Float16* __restrict__ qf, const _Float16* __restrict__ kf,
    const _Float16* __restrict__ vf, _Float16* __restrict__ o) {
  __shared__ _Float16 sK[32 * 72];      // [key][d] 32x64 (+8 pad)
  __shared__ _Float16 sVt[64 * 40];     // [d][key] transposed 64x32 (+8 pad)
  __shared__ _Float16 sP[8 * 16 * 40];  // per-wave 16x32 P tile (+8 pad)
  const int t = threadIdx.x;
  const int wave = t >> 5, lane = t & 31, lid = lane & 15, hl = lane >> 4;
  const int bh = blockIdx.x;
  const int bb = bh / HEADS, hh = bh % HEADS;
  const int q0 = blockIdx.y * 128 + wave * 16;
  const _Float16* qb = qf + (size_t)bh * SEQ * HDIM;
  const _Float16* kb = kf + (size_t)bh * SEQ * HDIM;
  const _Float16* vb = vf + (size_t)bh * SEQ * HDIM;

  // Q fragments (K-dim steps 0/1), wide 16B global loads, kept in registers.
  const _Float16* qrow = qb + (size_t)(q0 + lid) * HDIM;
  v16h qa0 = frag16(qrow + 8 * hl, qrow + 16 + 8 * hl);
  v16h qa1 = frag16(qrow + 32 + 8 * hl, qrow + 48 + 8 * hl);

  const v8f zero8 = {0.f, 0.f, 0.f, 0.f, 0.f, 0.f, 0.f, 0.f};
  v8f acc[4];
  float mrow[8], lrow[8];
#pragma unroll
  for (int dc = 0; dc < 4; ++dc) acc[dc] = zero8;
#pragma unroll
  for (int r = 0; r < 8; ++r) { mrow[r] = -__builtin_inff(); lrow[r] = 0.f; }

  const int skey = t >> 3, sd = (t & 7) * 8;  // K stage: 16B per thread
  const int vd = t & 63, vk = (t >> 6) * 8;   // V stage: transpose
  _Float16* myP = sP + wave * 16 * 40;

  for (int c = 0; c < SEQ / 32; ++c) {
    const int kbase = c * 32;
    __syncthreads();
    {  // async K tile: per-lane 16B global -> LDS, tracked by ASYNCcnt
      unsigned ldsOff = (unsigned)(uintptr_t)(sK + skey * 72 + sd);
      const _Float16* gsrc = kb + (size_t)(kbase + skey) * HDIM + sd;
      asm volatile("global_load_async_to_lds_b128 %0, %1, off" ::"v"(ldsOff),
                   "v"(gsrc)
                   : "memory");
    }
    {  // V tile transposed: per-j lane-coalesced f16 reads, 16B LDS store
      v8h h;
#pragma unroll
      for (int j = 0; j < 8; ++j) h[j] = vb[(size_t)(kbase + vk + j) * HDIM + vd];
      *(v8h*)(sVt + vd * 40 + vk) = h;
    }
    if (c + 1 < SEQ / 32) {  // -> global_prefetch_b8
      __builtin_prefetch(kb + (size_t)(kbase + 32 + skey) * HDIM + sd, 0, 3);
      __builtin_prefetch(vb + (size_t)(kbase + 32 + vk) * HDIM + vd, 0, 3);
    }
    asm volatile("s_wait_asynccnt 0x0" ::: "memory");
    __syncthreads();

    // S = Q @ K^T : tiles (keys 0-15, 16-31); K-dim 64 = 2 wmma each.
    v8f s0 = zero8, s1 = zero8;
    s0 = wmma_f16(qa0, bfrag(sK, 72, lid, hl, 0), s0);
    s0 = wmma_f16(qa1, bfrag(sK, 72, lid, hl, 32), s0);
    s1 = wmma_f16(qa0, bfrag(sK, 72, lid + 16, hl, 0), s1);
    s1 = wmma_f16(qa1, bfrag(sK, 72, lid + 16, hl, 32), s1);

    // Online softmax: row r+8*hl lives in VGPR r across this half's 16 lanes.
#pragma unroll
    for (int r = 0; r < 8; ++r) {
      float mx = fmaxf(s0[r], s1[r]);
      mx = fmaxf(mx, __shfl_xor(mx, 1, 32));
      mx = fmaxf(mx, __shfl_xor(mx, 2, 32));
      mx = fmaxf(mx, __shfl_xor(mx, 4, 32));
      mx = fmaxf(mx, __shfl_xor(mx, 8, 32));
      float newm = fmaxf(mrow[r], mx);
      float corr = __expf(mrow[r] - newm);
      float p0 = __expf(s0[r] - newm);
      float p1 = __expf(s1[r] - newm);
      float rs = p0 + p1;
      rs += __shfl_xor(rs, 1, 32);
      rs += __shfl_xor(rs, 2, 32);
      rs += __shfl_xor(rs, 4, 32);
      rs += __shfl_xor(rs, 8, 32);
      lrow[r] = lrow[r] * corr + rs;
      mrow[r] = newm;
#pragma unroll
      for (int dc = 0; dc < 4; ++dc) acc[dc][r] *= corr;
      // stash P (f16) for re-layout into an A-fragment (wave-private tile;
      // LDS ops are in-order within a wave)
      myP[(r + 8 * hl) * 40 + lid] = (_Float16)p0;
      myP[(r + 8 * hl) * 40 + lid + 16] = (_Float16)p1;
    }

    v16h pa = afrag(myP, 40, lid, hl, 0);
#pragma unroll
    for (int dc = 0; dc < 4; ++dc) {
      // sVt is [d][key] => bfrag with col=d reads contiguous keys (K-dim)
      acc[dc] = wmma_f16(pa, bfrag(sVt, 40, dc * 16 + lid, hl, 0), acc[dc]);
    }
  }

  // out[b, q0+row, hh*64 + d] = acc/l  (f16, [B,N,C] layout for proj GEMM)
  const size_t obase = ((size_t)bb * SEQ + q0) * CDIM + hh * HDIM;
#pragma unroll
  for (int r = 0; r < 8; ++r) {
    int row = r + 8 * hl;
    float inv = 1.0f / lrow[r];
#pragma unroll
    for (int dc = 0; dc < 4; ++dc)
      o[obase + (size_t)row * CDIM + dc * 16 + lid] =
          (_Float16)(acc[dc][r] * inv);
  }
}

// ---------------------------------------------------------------------------
// Kernel 3: out = attn_out(f16) @ w_proj + b_proj (f32 out).
// Block tile 128x64, 2x2 tiles per wave. Grid: (768/64, 4096/128).
// ---------------------------------------------------------------------------
__global__ __launch_bounds__(256) void proj_gemm_kernel(
    const _Float16* __restrict__ ain, const float* __restrict__ w,
    const float* __restrict__ bias, float* __restrict__ out) {
  __shared__ _Float16 sA[128 * 72];
  __shared__ _Float16 sB[64 * 72];  // [col][k]
  const int t = threadIdx.x;
  const int wave = t >> 5, lane = t & 31, lid = lane & 15, hl = lane >> 4;
  const int m0 = blockIdx.y * 128;
  const int n0 = blockIdx.x * 64;
  const int pr = wave & 3;
  const int pc = wave >> 2;
  const int ar = t >> 1, ak = (t & 1) * 32;
  const int col = t & 63, kc = (t >> 6) * 16;
  const v8f zero8 = {0.f, 0.f, 0.f, 0.f, 0.f, 0.f, 0.f, 0.f};
  v8f acc[2][2];
#pragma unroll
  for (int rr = 0; rr < 2; ++rr)
#pragma unroll
    for (int cc = 0; cc < 2; ++cc) acc[rr][cc] = zero8;

  for (int k0 = 0; k0 < CDIM; k0 += 64) {
    __syncthreads();
    {  // A already f16: four 16B loads + stores
#pragma unroll
      for (int q = 0; q < 4; ++q)
        *(v8h*)(sA + ar * 72 + ak + 8 * q) =
            *(const v8h*)(ain + (size_t)(m0 + ar) * CDIM + k0 + ak + 8 * q);
      if (k0 + 64 < CDIM)
        __builtin_prefetch(ain + (size_t)(m0 + ar) * CDIM + k0 + 64 + ak, 0, 3);
    }
    {  // B transposed stage
#pragma unroll
      for (int q = 0; q < 2; ++q) {
        v8h h;
#pragma unroll
        for (int j = 0; j < 8; ++j)
          h[j] = (_Float16)w[(size_t)(k0 + kc + 8 * q + j) * CDIM + n0 + col];
        *(v8h*)(sB + col * 72 + kc + 8 * q) = h;
      }
      if (k0 + 64 < CDIM)
        __builtin_prefetch(w + (size_t)(k0 + 64 + kc) * CDIM + n0 + col, 0, 3);
    }
    __syncthreads();
    v16h af[2][2], bf[2][2];
#pragma unroll
    for (int rr = 0; rr < 2; ++rr)
#pragma unroll
      for (int ks = 0; ks < 2; ++ks)
        af[rr][ks] = afrag(sA + (pr * 32 + rr * 16) * 72, 72, lid, hl, 32 * ks);
#pragma unroll
    for (int cc = 0; cc < 2; ++cc)
#pragma unroll
      for (int ks = 0; ks < 2; ++ks)
        bf[cc][ks] = bfrag(sB, 72, pc * 32 + cc * 16 + lid, hl, 32 * ks);
#pragma unroll
    for (int rr = 0; rr < 2; ++rr)
#pragma unroll
      for (int cc = 0; cc < 2; ++cc) {
        acc[rr][cc] = wmma_f16(af[rr][0], bf[cc][0], acc[rr][cc]);
        acc[rr][cc] = wmma_f16(af[rr][1], bf[cc][1], acc[rr][cc]);
      }
  }

#pragma unroll
  for (int rr = 0; rr < 2; ++rr) {
    const int rowbase = m0 + (pr * 2 + rr) * 16;
#pragma unroll
    for (int cc = 0; cc < 2; ++cc) {
      const int colbase = n0 + (pc * 2 + cc) * 16;
      const float bv = bias[colbase + lid];
      float* tbase = out + (size_t)rowbase * CDIM + colbase + lid;
#pragma unroll
      for (int r = 0; r < 8; ++r)
        tbase[(size_t)(r + 8 * hl) * CDIM] = acc[rr][cc][r] + bv;
    }
  }
}

extern "C" void kernel_launch(void* const* d_in, const int* in_sizes, int n_in,
                              void* d_out, int out_size, void* d_ws,
                              size_t ws_size, hipStream_t stream) {
  (void)in_sizes; (void)n_in; (void)out_size; (void)ws_size;
  const float* x      = (const float*)d_in[0];
  const float* w_qkv  = (const float*)d_in[1];
  const float* b_qkv  = (const float*)d_in[2];
  const float* w_proj = (const float*)d_in[3];
  const float* b_proj = (const float*)d_in[4];
  float* out = (float*)d_out;

  const size_t per = (size_t)BATCH * HEADS * SEQ * HDIM;  // 3,145,728 f16
  _Float16* qf = (_Float16*)d_ws;
  _Float16* kf = qf + per;
  _Float16* vf = kf + per;
  _Float16* ao = vf + per;  // [B*N, C] f16 attention output

  qkv_gemm_kernel<<<dim3((3 * CDIM) / 64, (BATCH * SEQ) / 128), 256, 0, stream>>>(
      x, w_qkv, b_qkv, qf, kf, vf);
  attn_kernel<<<dim3(BATCH * HEADS, SEQ / 128), 256, 0, stream>>>(qf, kf, vf, ao);
  proj_gemm_kernel<<<dim3(CDIM / 64, (BATCH * SEQ) / 128), 256, 0, stream>>>(
      ao, w_proj, b_proj, out);
}
